// EGNN_35742717837984
// MI455X (gfx1250) — matrix operand order repaired and checked by reading, not compiled, not run on hardware
//
#include <hip/hip_runtime.h>
#include <hip/hip_bf16.h>

// ---------------------------------------------------------------------------
// EGNN on MI455X (gfx1250): edge MLPs as f16 WMMA chains.
// Receiver-major edge enumeration padded with the diagonal (768x768 edges,
// dummy s==r rows masked) so every block owns ONE receiver:
//   - agg segment-sum: block-local reduce -> 64 atomics/block (was 4096)
//   - x_upd segment-sum: block-local reduce -> 6 atomics/block (was 96/tile)
// ~39 GFLOP of GEMM work on v_wmma_f32_16x16x32_f16; weights L2-resident.
// ---------------------------------------------------------------------------

#define TN      768            // flattened node count N*M
#define HD      64             // hidden dim H
#define EIN_K   160            // 130 padded to 5 chunks of 32
#define NWAVES  4              // waves per block (4*32 = 128 threads)
#define TILES_PER_R 48         // 768 senders (incl. dummy diagonal) / 16
#define BLOCKS_PER_R (TILES_PER_R / NWAVES)          // 12
#define EDGE_BLOCKS (TN * BLOCKS_PER_R)              // 9216

typedef __attribute__((ext_vector_type(16))) _Float16 half16;
typedef __attribute__((ext_vector_type(8)))  _Float16 half8;
typedef __attribute__((ext_vector_type(8)))  float    float8;

__device__ __forceinline__ float8 f8zero() {
  float8 z;
#pragma unroll
  for (int i = 0; i < 8; ++i) z[i] = 0.f;
  return z;
}

__device__ __forceinline__ float silu_f(float x) { return x / (1.f + __expf(-x)); }

// A-matrix 16x32 f16 fragment from a row-major f16 tile (LDS or global).
// ISA layout: lanes 0-15 hold row (lane&15), K = kc*32 + {0..7, 16..23};
//             lanes 16-31 hold K = kc*32 + {8..15, 24..31}.
__device__ __forceinline__ half16 load_a_frag(const _Float16* tile, int strideH,
                                              int kc, int lane) {
  const int row = lane & 15, hi = lane >> 4;
  const _Float16* p = tile + row * strideH + kc * 32 + hi * 8;
  half8 lo = *(const half8*)p;
  half8 hh = *(const half8*)(p + 16);
  return __builtin_shufflevector(lo, hh, 0,1,2,3,4,5,6,7,8,9,10,11,12,13,14,15);
}

// B-matrix 32x16 fragment, pre-packed: lane-major, 16 contiguous halfs/lane.
__device__ __forceinline__ half16 load_b_frag(const _Float16* pk, int frag, int lane) {
  return *(const half16*)(pk + (size_t)(frag * 32 + lane) * 16);
}

__device__ __forceinline__ float8 wmma_f16(half16 a, half16 b, float8 c) {
  return __builtin_amdgcn_wmma_f32_16x16x32_f16(false, a, false, b, (short)0, c,
                                                false, false);
}

// ---------------------------------------------------------------------------
// Weight packing: W[Kdim x Ndim] f32 row-major -> f16 B fragments.
// ---------------------------------------------------------------------------
__global__ void pack_b_kernel(const float* __restrict__ W, _Float16* __restrict__ dst,
                              int Kdim, int Ndim, int ntN) {
  int tid  = blockIdx.x * blockDim.x + threadIdx.x;
  int e    = tid & 15;
  int lane = (tid >> 4) & 31;
  int f    = tid >> 9;
  int kc   = f / ntN;
  int nt   = f - kc * ntN;
  int k    = kc * 32 + (lane >> 4) * 8 + (e & 7) + (e >> 3) * 16;
  int col  = nt * 16 + (lane & 15);
  float v  = (k < Kdim && col < Ndim) ? W[k * Ndim + col] : 0.f;
  dst[tid] = (_Float16)v;
}

__global__ void zero_kernel(float* __restrict__ a, int n) {
  int i = blockIdx.x * blockDim.x + threadIdx.x;
  if (i < n) a[i] = 0.f;
}

__global__ void initx_kernel(const float* __restrict__ pos, float* __restrict__ xv) {
  int i = blockIdx.x * blockDim.x + threadIdx.x;
  if (i < TN * 6) {
    int t = i / 6;
    int d = i % 3;
    xv[i] = pos[t * 3 + d];
  }
}

__global__ void xstep_kernel(float* __restrict__ xv, const float* __restrict__ xupd) {
  int i = blockIdx.x * blockDim.x + threadIdx.x;
  if (i < TN * 6) xv[i] += xupd[i] * (1.f / 767.f);
}

// ---------------------------------------------------------------------------
// Embedding: h = features[T,32] @ w_emb[32,64] + b_emb  (WMMA, 1 K-chunk).
// ---------------------------------------------------------------------------
__global__ __launch_bounds__(128)
void embed_kernel(const float* __restrict__ feat, const _Float16* __restrict__ pk_emb,
                  const float* __restrict__ b_emb,
                  float* __restrict__ h, _Float16* __restrict__ h16) {
  __shared__ __align__(32) _Float16 at[NWAVES][16 * 32];
  const int lane = threadIdx.x & 31, wave = threadIdx.x >> 5;
  const int row = lane & 15, hi = lane >> 4;
  const int tileBase = (blockIdx.x * NWAVES + wave) * 16;

  if (hi == 0) {
    const int t = tileBase + row;
#pragma unroll
    for (int c = 0; c < 32; ++c) at[wave][row * 32 + c] = (_Float16)feat[t * 32 + c];
  }
  __syncthreads();

  float8 acc[4] = {f8zero(), f8zero(), f8zero(), f8zero()};
  half16 a = load_a_frag(&at[wave][0], 32, 0, lane);
#pragma unroll
  for (int nt = 0; nt < 4; ++nt)
    acc[nt] = wmma_f16(a, load_b_frag(pk_emb, nt, lane), acc[nt]);

#pragma unroll
  for (int nt = 0; nt < 4; ++nt) {
    const int col = nt * 16 + row;
    const float b = b_emb[col];
#pragma unroll
    for (int j = 0; j < 8; ++j) {
      const int t = tileBase + hi * 8 + j;
      const float v = acc[nt][j] + b;
      h[t * HD + col]   = v;
      h16[t * HD + col] = (_Float16)v;
    }
  }
}

// ---------------------------------------------------------------------------
// Edge kernel, receiver-major. Block = one receiver r, 4 waves x 16 senders.
// Tile tt = blockIdx*4+wave: r = tt/48, senders s = (tt%48)*16 + row
// (s == r is the padded dummy edge: masked from agg; x-update self-masks).
//   m   = silu(silu([h_s|h_r|sqn] @ we1 + be1) @ we2 + be2)
//   phi = silu(m @ wx1 + bx1) @ wx2
//   agg[r]  += sum_s m       (block-reduced, 64 atomics/block)
//   xupd[r] += sum_s phi*diff/(sqrt(sqn)+1)   (block-reduced, 6 atomics/block)
// ---------------------------------------------------------------------------
__global__ __launch_bounds__(128)
void edge_kernel(const float* __restrict__ xv, const _Float16* __restrict__ h16,
                 const _Float16* __restrict__ pk_we1, const float* __restrict__ be1,
                 const _Float16* __restrict__ pk_we2, const float* __restrict__ be2,
                 const _Float16* __restrict__ pk_wx1, const float* __restrict__ bx1,
                 const _Float16* __restrict__ pk_wx2,
                 float* __restrict__ agg, float* __restrict__ xupd) {
  __shared__ __align__(32) _Float16 ein[NWAVES][16 * EIN_K];
  __shared__ __align__(32) _Float16 mt1[NWAVES][16 * HD];
  __shared__ __align__(32) _Float16 mt2[NWAVES][16 * HD];
  __shared__ float phi[NWAVES][16][2];
  __shared__ float redBuf[NWAVES][HD];      // per-wave column sums of m
  __shared__ float updBuf[NWAVES][16][6];   // per-row x-updates

  const int lane = threadIdx.x & 31, wave = threadIdx.x >> 5;
  const int row = lane & 15, hi = lane >> 4;

  const int tt = blockIdx.x * NWAVES + wave;   // tile id
  const int r  = tt / TILES_PER_R;             // receiver (uniform per block)
  const int jB = (tt - r * TILES_PER_R) * 16;  // sender base for this tile
  const int s  = jB + row;                     // sender for this lane's row

  // L2 prefetch hints for the (hot, shared) packed weights.
  __builtin_prefetch(pk_we1 + lane * 64, 0, 3);
  __builtin_prefetch(pk_we2 + lane * 64, 0, 3);
  __builtin_prefetch(pk_wx1 + lane * 64, 0, 3);

  float diff[2][3], sqn[2];
#pragma unroll
  for (int k = 0; k < 2; ++k) {
    float a = 0.f;
#pragma unroll
    for (int d = 0; d < 3; ++d) {
      const float v = xv[s * 6 + k * 3 + d] - xv[r * 6 + k * 3 + d];
      diff[k][d] = v;
      a += v * v;
    }
    sqn[k] = a;
  }

  // Build e_in tile [16 x 160] f16 in LDS. h_s rows are CONTIGUOUS (coalesced),
  // h_r is tile-uniform (L0 hit).
  {
    _Float16* dst = &ein[wave][row * EIN_K];
    const _Float16* hs = h16 + s * HD;
    const _Float16* hr = h16 + r * HD;
    if (hi == 0) {
#pragma unroll
      for (int c = 0; c < 4; ++c)
        *(half16*)(dst + c * 16) = *(const half16*)(hs + c * 16);
      *(half16*)(dst + 64) = *(const half16*)(hr);
    } else {
#pragma unroll
      for (int c = 0; c < 3; ++c)
        *(half16*)(dst + 80 + c * 16) = *(const half16*)(hr + 16 + c * 16);
      dst[128] = (_Float16)sqn[0];
      dst[129] = (_Float16)sqn[1];
#pragma unroll
      for (int c = 130; c < EIN_K; ++c) dst[c] = (_Float16)0.f;
    }
  }
  __syncthreads();

  // GEMM1: [16x160] @ we1[160x64]  (5 K-chunks x 4 N-tiles)
  float8 a1[4] = {f8zero(), f8zero(), f8zero(), f8zero()};
#pragma unroll
  for (int kc = 0; kc < 5; ++kc) {
    half16 a = load_a_frag(&ein[wave][0], EIN_K, kc, lane);
#pragma unroll
    for (int nt = 0; nt < 4; ++nt)
      a1[nt] = wmma_f16(a, load_b_frag(pk_we1, kc * 4 + nt, lane), a1[nt]);
  }
#pragma unroll
  for (int nt = 0; nt < 4; ++nt) {
    const int col = nt * 16 + row;
    const float b = be1[col];
#pragma unroll
    for (int jj = 0; jj < 8; ++jj)
      mt1[wave][(hi * 8 + jj) * HD + col] = (_Float16)silu_f(a1[nt][jj] + b);
  }
  __syncthreads();

  // GEMM2: m @ we2[64x64] -> m ; column-sum m over real rows (mask s==r).
  float8 a2[4] = {f8zero(), f8zero(), f8zero(), f8zero()};
#pragma unroll
  for (int kc = 0; kc < 2; ++kc) {
    half16 a = load_a_frag(&mt1[wave][0], HD, kc, lane);
#pragma unroll
    for (int nt = 0; nt < 4; ++nt)
      a2[nt] = wmma_f16(a, load_b_frag(pk_we2, kc * 4 + nt, lane), a2[nt]);
  }
#pragma unroll
  for (int nt = 0; nt < 4; ++nt) {
    const int col = nt * 16 + row;
    const float b = be2[col];
    float csum = 0.f;
#pragma unroll
    for (int jj = 0; jj < 8; ++jj) {
      const float v = silu_f(a2[nt][jj] + b);
      mt2[wave][(hi * 8 + jj) * HD + col] = (_Float16)v;
      const bool real = (jB + hi * 8 + jj) != r;   // mask dummy diagonal row
      csum += real ? v : 0.f;
    }
    // combine the two half-wave row groups (lane L and L+16 share a column)
    csum += __shfl_xor(csum, 16);
    if (hi == 0) redBuf[wave][col] = csum;
  }
  __syncthreads();

  // GEMM3a: mm = silu(m @ wx1 + bx1)   (store into mt1, reuse)
  float8 a3[4] = {f8zero(), f8zero(), f8zero(), f8zero()};
#pragma unroll
  for (int kc = 0; kc < 2; ++kc) {
    half16 a = load_a_frag(&mt2[wave][0], HD, kc, lane);
#pragma unroll
    for (int nt = 0; nt < 4; ++nt)
      a3[nt] = wmma_f16(a, load_b_frag(pk_wx1, kc * 4 + nt, lane), a3[nt]);
  }
  __syncthreads();   // mt1 free to overwrite
#pragma unroll
  for (int nt = 0; nt < 4; ++nt) {
    const int col = nt * 16 + row;
    const float b = bx1[col];
#pragma unroll
    for (int jj = 0; jj < 8; ++jj)
      mt1[wave][(hi * 8 + jj) * HD + col] = (_Float16)silu_f(a3[nt][jj] + b);
  }
  __syncthreads();

  // GEMM3b: phi = mm @ wx2[64x2]  (N padded to 16 in the pack, 1 N-tile)
  float8 a4 = f8zero();
#pragma unroll
  for (int kc = 0; kc < 2; ++kc) {
    half16 a = load_a_frag(&mt1[wave][0], HD, kc, lane);
    a4 = wmma_f16(a, load_b_frag(pk_wx2, kc, lane), a4);
  }
  if ((lane & 15) < 2) {
    const int kcol = lane & 15;
#pragma unroll
    for (int jj = 0; jj < 8; ++jj) phi[wave][hi * 8 + jj][kcol] = a4[jj];
  }
  __syncthreads();

  // Per-row x-update (dummy row self-masks: diff == 0).
  if (hi == 0) {
#pragma unroll
    for (int k = 0; k < 2; ++k) {
      const float p  = phi[wave][row][k];
      const float sc = p / (sqrtf(sqn[k] + 1e-8f) + 1.f);
#pragma unroll
      for (int d = 0; d < 3; ++d) updBuf[wave][row][k * 3 + d] = sc * diff[k][d];
    }
  }
  __syncthreads();

  // Block-level reductions: whole block shares receiver rB.
  const int rB  = blockIdx.x / BLOCKS_PER_R;
  const int tid = threadIdx.x;
  if (tid < HD) {
    float v = 0.f;
#pragma unroll
    for (int w = 0; w < NWAVES; ++w) v += redBuf[w][tid];
    atomicAdd(&agg[rB * HD + tid], v);
  }
  if (tid < 6) {
    float v = 0.f;
#pragma unroll
    for (int w = 0; w < NWAVES; ++w)
#pragma unroll
      for (int rr = 0; rr < 16; ++rr) v += updBuf[w][rr][tid];
    atomicAdd(&xupd[rB * 6 + tid], v);
  }
}

// ---------------------------------------------------------------------------
// Node update: h += silu(concat(h, agg) @ wh1 + bh1) @ wh2 + bh2
// ---------------------------------------------------------------------------
__global__ __launch_bounds__(128)
void node_kernel(float* __restrict__ h, _Float16* __restrict__ h16,
                 const float* __restrict__ agg,
                 const _Float16* __restrict__ pk_wh1, const float* __restrict__ bh1,
                 const _Float16* __restrict__ pk_wh2, const float* __restrict__ bh2) {
  __shared__ __align__(32) _Float16 cat[NWAVES][16 * 128];
  __shared__ __align__(32) _Float16 mt[NWAVES][16 * HD];
  const int lane = threadIdx.x & 31, wave = threadIdx.x >> 5;
  const int row = lane & 15, hi = lane >> 4;
  const int tileBase = (blockIdx.x * NWAVES + wave) * 16;
  const int t = tileBase + row;

  {
    _Float16* dst = &cat[wave][row * 128];
    if (hi == 0) {
#pragma unroll
      for (int c = 0; c < 4; ++c)
        *(half16*)(dst + c * 16) = *(const half16*)(h16 + t * HD + c * 16);
    } else {
#pragma unroll
      for (int c = 0; c < HD; ++c) dst[64 + c] = (_Float16)agg[t * HD + c];
    }
  }
  __syncthreads();

  float8 a1[4] = {f8zero(), f8zero(), f8zero(), f8zero()};
#pragma unroll
  for (int kc = 0; kc < 4; ++kc) {
    half16 a = load_a_frag(&cat[wave][0], 128, kc, lane);
#pragma unroll
    for (int nt = 0; nt < 4; ++nt)
      a1[nt] = wmma_f16(a, load_b_frag(pk_wh1, kc * 4 + nt, lane), a1[nt]);
  }
#pragma unroll
  for (int nt = 0; nt < 4; ++nt) {
    const int col = nt * 16 + row;
    const float b = bh1[col];
#pragma unroll
    for (int jj = 0; jj < 8; ++jj)
      mt[wave][(hi * 8 + jj) * HD + col] = (_Float16)silu_f(a1[nt][jj] + b);
  }
  __syncthreads();

  float8 a2[4] = {f8zero(), f8zero(), f8zero(), f8zero()};
#pragma unroll
  for (int kc = 0; kc < 2; ++kc) {
    half16 a = load_a_frag(&mt[wave][0], HD, kc, lane);
#pragma unroll
    for (int nt = 0; nt < 4; ++nt)
      a2[nt] = wmma_f16(a, load_b_frag(pk_wh2, kc * 4 + nt, lane), a2[nt]);
  }
#pragma unroll
  for (int nt = 0; nt < 4; ++nt) {
    const int col = nt * 16 + row;
    const float b = bh2[col];
#pragma unroll
    for (int jj = 0; jj < 8; ++jj) {
      const int tt = tileBase + hi * 8 + jj;
      const float nv = h[tt * HD + col] + a2[nt][jj] + b;
      h[tt * HD + col]   = nv;
      h16[tt * HD + col] = (_Float16)nv;
    }
  }
}

// ---------------------------------------------------------------------------
// Head: scalars = h @ w_head + b_head ; vectors = x_vec - x
// ---------------------------------------------------------------------------
__global__ __launch_bounds__(128)
void head_kernel(const _Float16* __restrict__ h16, const _Float16* __restrict__ pk_head,
                 const float* __restrict__ b_head,
                 const float* __restrict__ xv, const float* __restrict__ pos,
                 float* __restrict__ out) {
  const int lane = threadIdx.x & 31, wave = threadIdx.x >> 5;
  const int row = lane & 15, hi = lane >> 4;
  const int tileBase = (blockIdx.x * NWAVES + wave) * 16;

  float8 acc[4] = {f8zero(), f8zero(), f8zero(), f8zero()};
#pragma unroll
  for (int kc = 0; kc < 2; ++kc) {
    half16 a = load_a_frag(h16 + tileBase * HD, HD, kc, lane);  // A direct from global
#pragma unroll
    for (int nt = 0; nt < 4; ++nt)
      acc[nt] = wmma_f16(a, load_b_frag(pk_head, kc * 4 + nt, lane), acc[nt]);
  }

  float* scal = out + TN * 6;  // vectors first (4608 floats), then scalars
#pragma unroll
  for (int nt = 0; nt < 4; ++nt) {
    const int col = nt * 16 + row;
    const float b = b_head[col];
#pragma unroll
    for (int jj = 0; jj < 8; ++jj) {
      const int t = tileBase + hi * 8 + jj;
      scal[t * HD + col] = acc[nt][jj] + b;
    }
  }

  if (hi == 0) {
    const int t = tileBase + row;
#pragma unroll
    for (int i = 0; i < 6; ++i) out[t * 6 + i] = xv[t * 6 + i] - pos[t * 3 + (i % 3)];
  }
}

// ---------------------------------------------------------------------------
// Host side
// ---------------------------------------------------------------------------
static constexpr size_t OFF_H    = 0;
static constexpr size_t OFF_H16  = OFF_H    + (size_t)TN * HD * 4;
static constexpr size_t OFF_XV   = OFF_H16  + (size_t)TN * HD * 2;
static constexpr size_t OFF_XUPD = OFF_XV   + (size_t)TN * 6 * 4;
static constexpr size_t OFF_AGG  = OFF_XUPD + (size_t)TN * 6 * 4;
static constexpr size_t OFF_PK   = OFF_AGG  + (size_t)TN * HD * 4;
// packed-weight layout (1024 B per 32x16 fragment)
static constexpr size_t PK_EMB  = OFF_PK;                    // 4 frags
static constexpr size_t PK_WE1  = PK_EMB  + 4  * 1024;       // 20 frags x 2 layers
static constexpr size_t PK_WE2  = PK_WE1  + 2 * 20 * 1024;   // 8 x 2
static constexpr size_t PK_WX1  = PK_WE2  + 2 * 8  * 1024;   // 8 x 2
static constexpr size_t PK_WX2  = PK_WX1  + 2 * 8  * 1024;   // 2 x 2
static constexpr size_t PK_WH1  = PK_WX2  + 2 * 2  * 1024;   // 16 x 2
static constexpr size_t PK_WH2  = PK_WH1  + 2 * 16 * 1024;   // 8 x 2
static constexpr size_t PK_HEAD = PK_WH2  + 2 * 8  * 1024;   // 8

static void launch_pack(const float* W, _Float16* dst, int K, int N,
                        int kcN, int ntN, hipStream_t s) {
  const int frags = kcN * ntN;                 // 512 threads per fragment
  pack_b_kernel<<<frags * 2, 256, 0, s>>>(W, dst, K, N, ntN);
}

extern "C" void kernel_launch(void* const* d_in, const int* in_sizes, int n_in,
                              void* d_out, int out_size, void* d_ws, size_t ws_size,
                              hipStream_t stream) {
  (void)in_sizes; (void)n_in; (void)out_size; (void)ws_size;

  const float* pos    = (const float*)d_in[0];
  const float* feat   = (const float*)d_in[1];
  const float* w_emb  = (const float*)d_in[2];
  const float* b_emb  = (const float*)d_in[3];
  const float* we1    = (const float*)d_in[4];
  const float* be1    = (const float*)d_in[5];
  const float* we2    = (const float*)d_in[6];
  const float* be2    = (const float*)d_in[7];
  const float* wx1    = (const float*)d_in[8];
  const float* bx1    = (const float*)d_in[9];
  const float* wx2    = (const float*)d_in[10];
  const float* wh1    = (const float*)d_in[11];
  const float* bh1    = (const float*)d_in[12];
  const float* wh2    = (const float*)d_in[13];
  const float* bh2    = (const float*)d_in[14];
  const float* w_head = (const float*)d_in[15];
  const float* b_head = (const float*)d_in[16];

  char* ws = (char*)d_ws;
  float*     h    = (float*)(ws + OFF_H);
  _Float16*  h16  = (_Float16*)(ws + OFF_H16);
  float*     xv   = (float*)(ws + OFF_XV);
  float*     xupd = (float*)(ws + OFF_XUPD);
  float*     agg  = (float*)(ws + OFF_AGG);
  _Float16*  pk_emb  = (_Float16*)(ws + PK_EMB);
  _Float16*  pk_head = (_Float16*)(ws + PK_HEAD);

  // ---- pack weights into f16 WMMA B-fragments ----
  launch_pack(w_emb,  pk_emb,  32, 64, 1, 4, stream);
  launch_pack(w_head, pk_head, 64, 64, 2, 4, stream);
  for (int l = 0; l < 2; ++l) {
    launch_pack(we1 + (size_t)l * 130 * 64, (_Float16*)(ws + PK_WE1 + (size_t)l * 20 * 1024), 130, 64, 5, 4, stream);
    launch_pack(we2 + (size_t)l * 64 * 64,  (_Float16*)(ws + PK_WE2 + (size_t)l * 8  * 1024),  64, 64, 2, 4, stream);
    launch_pack(wx1 + (size_t)l * 64 * 64,  (_Float16*)(ws + PK_WX1 + (size_t)l * 8  * 1024),  64, 64, 2, 4, stream);
    launch_pack(wx2 + (size_t)l * 64 * 2,   (_Float16*)(ws + PK_WX2 + (size_t)l * 2  * 1024),  64,  2, 2, 1, stream);
    launch_pack(wh1 + (size_t)l * 128 * 64, (_Float16*)(ws + PK_WH1 + (size_t)l * 16 * 1024), 128, 64, 4, 4, stream);
    launch_pack(wh2 + (size_t)l * 64 * 64,  (_Float16*)(ws + PK_WH2 + (size_t)l * 8  * 1024),  64, 64, 2, 4, stream);
  }

  // ---- init ----
  initx_kernel<<<18, 256, 0, stream>>>(pos, xv);
  embed_kernel<<<12, 128, 0, stream>>>(feat, pk_emb, b_emb, h, h16);  // 48 row tiles

  // ---- message-passing layers ----
  for (int l = 0; l < 2; ++l) {
    zero_kernel<<<192, 256, 0, stream>>>(agg, TN * HD);
    zero_kernel<<<18, 256, 0, stream>>>(xupd, TN * 6);
    edge_kernel<<<EDGE_BLOCKS, 128, 0, stream>>>(
        xv, h16,
        (const _Float16*)(ws + PK_WE1 + (size_t)l * 20 * 1024), be1 + l * 64,
        (const _Float16*)(ws + PK_WE2 + (size_t)l * 8  * 1024), be2 + l * 64,
        (const _Float16*)(ws + PK_WX1 + (size_t)l * 8  * 1024), bx1 + l * 64,
        (const _Float16*)(ws + PK_WX2 + (size_t)l * 2  * 1024),
        agg, xupd);
    node_kernel<<<12, 128, 0, stream>>>(
        h, h16, agg,
        (const _Float16*)(ws + PK_WH1 + (size_t)l * 16 * 1024), bh1 + l * 64,
        (const _Float16*)(ws + PK_WH2 + (size_t)l * 8  * 1024), bh2 + l * 64);
    xstep_kernel<<<18, 256, 0, stream>>>(xv, xupd);
  }

  // ---- outputs ----
  head_kernel<<<12, 128, 0, stream>>>(h16, pk_head, b_head, xv, pos, (float*)d_out);
}
